// RNN_14173392077300
// MI455X (gfx1250) — compile-verified, matrix operand order checked by reading, not run on hardware
//
#include <hip/hip_runtime.h>
#include <hip/hip_bf16.h>

#define VOCAB  14
#define EMBED  10
#define HIDDEN 50
#define T_SEQ  2048
#define B_TOT  4096
#define KPAD   64      // hidden dim padded for WMMA K
#define LOG2E  1.4426950408889634f

typedef __attribute__((ext_vector_type(16))) _Float16 v16h;
typedef __attribute__((ext_vector_type(8)))  _Float16 v8h;
typedef __attribute__((ext_vector_type(8)))  float    v8f;

#if __has_builtin(__builtin_amdgcn_sched_barrier)
#define SCHED_FENCE() __builtin_amdgcn_sched_barrier(0)
#else
#define SCHED_FENCE()
#endif

__device__ __forceinline__ float fsig(float x) {
    // sigmoid(x) = 1 / (1 + 2^(-x*log2e))  -> v_exp_f32 + v_rcp_f32
    return __builtin_amdgcn_rcpf(1.0f + __builtin_amdgcn_exp2f(-LOG2E * x));
}

#if __has_builtin(__builtin_amdgcn_tanhf)
__device__ __forceinline__ float ftanh_(float x) { return __builtin_amdgcn_tanhf(x); }
#elif __has_builtin(__builtin_amdgcn_tanh_f32)
__device__ __forceinline__ float ftanh_(float x) { return __builtin_amdgcn_tanh_f32(x); }
#else
__device__ __forceinline__ float ftanh_(float x) {
    float e = __builtin_amdgcn_exp2f(2.0f * LOG2E * x);   // e^(2x)
    return 1.0f - 2.0f * __builtin_amdgcn_rcpf(e + 1.0f);
}
#endif

__global__ __launch_bounds__(32)
void lstm_wave_kernel(const int*   __restrict__ tokens,
                      const float* __restrict__ emb,
                      const float* __restrict__ W_ih,
                      const float* __restrict__ W_hh,
                      const float* __restrict__ b_ih,
                      const float* __restrict__ b_hh,
                      const float* __restrict__ W1,
                      const float* __restrict__ b1,
                      const float* __restrict__ W2,
                      const float* __restrict__ b2,
                      float*       __restrict__ out)
{
    // W_hh pre-swizzled into per-lane WMMA-B layout: [nt][kc][lane][16 f16] (32 KB)
    __shared__ __align__(32) _Float16 WB[16 * 2 * 32 * 16];
    // xgate+bias table: [tok][c16][nt] f32 (14 KB)
    __shared__ __align__(16) float    XG[VOCAB * 16 * 16];
    // h state f16, [M][K] row-major (2 KB)
    __shared__ __align__(32) _Float16 h16[16 * KPAD];

    const int lane = threadIdx.x;        // 0..31
    const int c16  = lane & 15;
    const int hlf  = lane >> 4;
    const int rowBase = blockIdx.x * 16;

    // ---- build xgate table: XG[tok][c][nt] = emb[tok] . W_ih[col] + b_ih + b_hh ----
    for (int idx = lane; idx < VOCAB * 256; idx += 32) {
        int tok = idx >> 8;
        int rem = idx & 255;
        int c   = rem >> 4, nt = rem & 15;
        int col = nt * 16 + c;           // padded gate column 0..255
        int q   = col >> 6, j = col & 63;
        float v = 0.0f;
        if (j < HIDDEN) {
            int row = q * HIDDEN + j;
            v = b_ih[row] + b_hh[row];
            for (int e = 0; e < EMBED; ++e)
                v += emb[tok * EMBED + e] * W_ih[row * EMBED + e];
        }
        XG[tok * 256 + c * 16 + nt] = v;
    }

    // ---- build WB: W_hh as f16 in WMMA B operand per-lane order ----
    for (int idx = lane; idx < 16 * 2 * 32 * 16; idx += 32) {
        int e  = idx & 15;
        int l  = (idx >> 4) & 31;
        int kc = (idx >> 9) & 1;
        int nt = idx >> 10;
        int N  = nt * 16 + (l & 15);     // gate column (padded)
        int lh = l >> 4;
        int p  = e >> 1, w = e & 1;
        int kb = (p < 4) ? (2 * p + 8 * lh) : (16 + 2 * (p - 4) + 8 * lh);
        int k  = kc * 32 + kb + w;       // hidden index
        int q  = N >> 6, j = N & 63;
        float v = 0.0f;
        if (j < HIDDEN && k < HIDDEN) v = W_hh[(q * HIDDEN + j) * HIDDEN + k];
        WB[idx] = (_Float16)v;
    }

    for (int idx = lane; idx < 16 * KPAD; idx += 32) h16[idx] = (_Float16)0.0f;
    __syncthreads();

    v8f   acc[16];
    v8f   zc = {};                        // shared zero C operand
    float cst[4][8];
#pragma unroll
    for (int jt = 0; jt < 4; ++jt)
#pragma unroll
        for (int r = 0; r < 8; ++r) cst[jt][r] = 0.0f;

    const int* tokRow = tokens + (size_t)(rowBase + c16) * T_SEQ;
    const int  Mbase  = 8 * hlf;

#pragma unroll 1
    for (int t = 0; t < T_SEQ; ++t) {
        int tok = tokRow[t];
        int tp = t + 192; if (tp > T_SEQ - 1) tp = T_SEQ - 1;
        __builtin_prefetch(&tokRow[tp], 0, 0);

        int tokr[8];
#pragma unroll
        for (int r = 0; r < 8; ++r) tokr[r] = __shfl(tok, Mbase + r, 32);

        // ---- A operands: h (f16) in WMMA A layout, M = c16, K per half ----
        const v8h* hrow = (const v8h*)&h16[c16 * KPAD];   // 8 groups of 8 f16
        v8h lo0 = hrow[hlf],     hi0 = hrow[2 + hlf];
        v8h lo1 = hrow[4 + hlf], hi1 = hrow[6 + hlf];
        v16h A0 = __builtin_shufflevector(lo0, hi0, 0,1,2,3,4,5,6,7,8,9,10,11,12,13,14,15);
        v16h A1 = __builtin_shufflevector(lo1, hi1, 0,1,2,3,4,5,6,7,8,9,10,11,12,13,14,15);

        // ---- recurrent GEMM: 16 N-tiles x 2 K-chunks, C starts at zero ----
#pragma unroll
        for (int nt = 0; nt < 16; ++nt) {
            const v16h* b0 = (const v16h*)&WB[((nt * 2 + 0) * 32 + lane) * 16];
            const v16h* b1v = (const v16h*)&WB[((nt * 2 + 1) * 32 + lane) * 16];
            acc[nt] = __builtin_amdgcn_wmma_f32_16x16x32_f16(
                false, A0, false, *b0, (short)0, zc, false, false);
            acc[nt] = __builtin_amdgcn_wmma_f32_16x16x32_f16(
                false, A1, false, *b1v, (short)0, acc[nt], false, false);
            if ((nt & 3) == 3) SCHED_FENCE();   // cap B-operand live range
        }

        // ---- pointwise LSTM cell; xgate contribution added here (no placement) ----
#pragma unroll
        for (int r = 0; r < 8; ++r) {
            const float4* xg = (const float4*)&XG[tokr[r] * 256 + c16 * 16];
            float4 xi = xg[0], xf = xg[1], xgv = xg[2], xo = xg[3];
            float xia[4] = {xi.x, xi.y, xi.z, xi.w};
            float xfa[4] = {xf.x, xf.y, xf.z, xf.w};
            float xga[4] = {xgv.x, xgv.y, xgv.z, xgv.w};
            float xoa[4] = {xo.x, xo.y, xo.z, xo.w};
            bool  m = (tokr[r] != 0);
            float hn[4];
#pragma unroll
            for (int jt = 0; jt < 4; ++jt) {
                float iv = fsig (acc[jt     ][r] + xia[jt]);
                float fv = fsig (acc[jt +  4][r] + xfa[jt]);
                float gv = ftanh_(acc[jt +  8][r] + xga[jt]);
                float ov = fsig (acc[jt + 12][r] + xoa[jt]);
                float cn = fv * cst[jt][r] + iv * gv;
                hn[jt] = ov * ftanh_(cn);
                cst[jt][r] = m ? cn : cst[jt][r];
            }
            if (m) {                     // masked rows keep old h automatically
#pragma unroll
                for (int jt = 0; jt < 4; ++jt)
                    h16[(Mbase + r) * KPAD + jt * 16 + c16] = (_Float16)hn[jt];
            }
        }
        __syncthreads();   // single-wave WG: compiler fence + cheap barrier
    }

    // ---- MLP head: hidden = h @ W1^T + b1 ; out = sigmoid(hidden @ W2^T + b2) ----
    if (lane < 16) {
        float s2 = b2[0];
        for (int k = 0; k < HIDDEN; ++k) {
            float s = b1[k];
            for (int j = 0; j < HIDDEN; ++j)
                s += (float)h16[lane * KPAD + j] * W1[k * HIDDEN + j];
            s2 += s * W2[k];
        }
        out[rowBase + lane] = fsig(s2);
    }
}

extern "C" void kernel_launch(void* const* d_in, const int* in_sizes, int n_in,
                              void* d_out, int out_size, void* d_ws, size_t ws_size,
                              hipStream_t stream) {
    const int*   tokens = (const int*)  d_in[0];
    const float* emb    = (const float*)d_in[1];
    const float* W_ih   = (const float*)d_in[2];
    const float* W_hh   = (const float*)d_in[3];
    const float* b_ih   = (const float*)d_in[4];
    const float* b_hh   = (const float*)d_in[5];
    const float* W1     = (const float*)d_in[6];
    const float* b1     = (const float*)d_in[7];
    const float* W2     = (const float*)d_in[8];
    const float* b2     = (const float*)d_in[9];
    float* out = (float*)d_out;

    hipLaunchKernelGGL(lstm_wave_kernel, dim3(B_TOT / 16), dim3(32), 0, stream,
                       tokens, emb, W_ih, W_hh, b_ih, b_hh, W1, b1, W2, b2, out);
}
